// OIMLossSMR_54760833024747
// MI455X (gfx1250) — matrix operand order repaired and checked by reading, not compiled, not run on hardware
//
#include <hip/hip_runtime.h>

// ---------------------------------------------------------------------------
// OIM loss (SMR) for MI455X / gfx1250.
// Fused bf16-WMMA GEMM + streaming log-sum-exp (logits bounded by |30 cos|<=30
// so a fixed -30 shift replaces online-max), exact-fp32 label logits,
// deterministic partial-sum reduction (no float atomics).
// B tiles are staged by the Tensor Data Mover (tensor_load_to_lds) from a
// pre-converted contiguous bf16 copy of [lut;cq]; the D# pad feature emits the
// bank-conflict-free 528-byte column stride directly. TDM for tile t+1
// overlaps WMMA compute of tile t; wave 0 drains TENSORcnt before the barrier.
// ---------------------------------------------------------------------------

typedef __attribute__((ext_vector_type(16))) __bf16 v16bf;
typedef __attribute__((ext_vector_type(2)))  __bf16 v2bf;
typedef __attribute__((ext_vector_type(8)))  float  v8f;
typedef __attribute__((ext_vector_type(4)))  unsigned int v4u;
typedef __attribute__((ext_vector_type(8)))  int    v8i;
typedef __attribute__((ext_vector_type(4)))  int    v4i;

#define NF        256       // feature dim (K)
#define NP        5532      // lut rows
#define NQ        5000      // cq rows
#define NC        10532     // NP + NQ
#define NROWS     8192      // N
#define IGNORE_ROI 5555     // roi value whose label is ignored
#define LOG2E30   43.28085122666891f   // 30 * log2(e)
#define LN2       0.6931471805599453f
#define ROWBLK    256       // rows per workgroup (8 waves x 2 x 16 rows)
#define NCB       8         // column blocks (grid.y)
#define NTILES    659       // ceil(10532 / 16)
#define TILES_PER_CB 83     // ceil(659 / 8)
#define CSTR      264       // bf16 elems per LDS column row (256 + 8 pad)
#define WBF_BYTES ((size_t)NC * NF * 2)   // 5,392,384 B

struct B32x8 { uint4 a, b; };

__device__ __forceinline__ unsigned pk2bf(float x, float y) {
  v2bf v; v.x = (__bf16)x; v.y = (__bf16)y;
  return __builtin_bit_cast(unsigned, v);
}

__device__ __forceinline__ v16bf cvt16(const float4& a, const float4& b,
                                       const float4& c, const float4& d) {
  v16bf v;
  v[0]  = (__bf16)a.x; v[1]  = (__bf16)a.y; v[2]  = (__bf16)a.z; v[3]  = (__bf16)a.w;
  v[4]  = (__bf16)b.x; v[5]  = (__bf16)b.y; v[6]  = (__bf16)b.z; v[7]  = (__bf16)b.w;
  v[8]  = (__bf16)c.x; v[9]  = (__bf16)c.y; v[10] = (__bf16)c.z; v[11] = (__bf16)c.w;
  v[12] = (__bf16)d.x; v[13] = (__bf16)d.y; v[14] = (__bf16)d.z; v[15] = (__bf16)d.w;
  return v;
}

// Load one wave's A tile (16 rows x 256 K, bf16) into 8 v16bf registers,
// laid out per the ISA 16-bit A layout.
__device__ __forceinline__ void load_A(v16bf A[8], const float* __restrict__ xrow,
                                       int hi) {
  #pragma unroll
  for (int kc = 0; kc < 8; ++kc) {
    const int b1 = kc * 32 + hi * 8;
    const float4* p = (const float4*)(xrow + b1);
    const float4* q = (const float4*)(xrow + b1 + 16);
    A[kc] = cvt16(p[0], p[1], q[0], q[1]);
  }
}

// Issue one TDM load: 16 rows x 256 bf16 tile of Wbf -> LDS, with LDS padding
// of 16 B after every 512 B row (=> 528 B column stride). Rows beyond
// rows_avail (tensor_dim1) are zero-filled by the TDM's OOB rule.
__device__ __forceinline__ void tdm_load_tile(const __bf16* gsrc,
                                              unsigned lds_off, int rows_avail) {
  const unsigned long long ga = (unsigned long long)(uintptr_t)gsrc;
  v4u g0;
  g0[0] = 1u;                                    // count=1, user descriptor
  g0[1] = lds_off;                               // LDS byte address
  g0[2] = (unsigned)(ga & 0xffffffffu);          // global_addr[31:0]
  g0[3] = (unsigned)(ga >> 32) | 0x80000000u;    // global_addr[56:32] | type=2
  v8i g1;
  g1[0] = (1 << 16)        // data_size = 2 bytes (bf16)
        | (1 << 20)        // pad_enable
        | (6 << 22)        // pad_interval: 128 DWORDs = 512 B (one row)
        | (3 << 25);       // pad_amount:   4 DWORDs = 16 B
  g1[1] = (NF & 0xffff) << 16;                   // tensor_dim0 = 256 (lo half)
  g1[2] = (rows_avail & 0xffff) << 16;           // tensor_dim1 lo (dim0 hi = 0)
  g1[3] = ((unsigned)rows_avail >> 16) | (NF << 16); // tensor_dim1 hi | tile_dim0
  g1[4] = 16;                                    // tile_dim1 = 16 (tile_dim2 = 0)
  g1[5] = NF;                                    // tensor_dim0_stride = 256
  g1[6] = 0;
  g1[7] = 0;
  const v4i gz = {0, 0, 0, 0};
#if __clang_major__ >= 23
  const v8i gz8 = {0, 0, 0, 0, 0, 0, 0, 0};
  __builtin_amdgcn_tensor_load_to_lds(g0, g1, gz, gz, gz8, 0);
#else
  __builtin_amdgcn_tensor_load_to_lds(g0, g1, gz, gz, 0);
#endif
}

// Kernel 0: convert W = [lut;cq] (fp32) to contiguous bf16 in workspace.
__global__ __launch_bounds__(256) void oim_cvt_w(
    const float* __restrict__ lut, const float* __restrict__ cq,
    __bf16* __restrict__ Wbf) {
  const size_t idx = ((size_t)blockIdx.x * 256 + threadIdx.x) * 16;
  if (idx >= (size_t)NC * NF) return;
  const int j = (int)(idx >> 8);        // row (16 | 256 => row-uniform chunk)
  const int k = (int)(idx & 255);
  const float* src = (j < NP) ? (lut + (size_t)j * NF + k)
                              : (cq + (size_t)(j - NP) * NF + k);
  const float4* p = (const float4*)src;
  float4 a = p[0], b = p[1], c = p[2], d = p[3];
  uint4 w0 = make_uint4(pk2bf(a.x, a.y), pk2bf(a.z, a.w),
                        pk2bf(b.x, b.y), pk2bf(b.z, b.w));
  uint4 w1 = make_uint4(pk2bf(c.x, c.y), pk2bf(c.z, c.w),
                        pk2bf(d.x, d.y), pk2bf(d.z, d.w));
  *(uint4*)(Wbf + idx)     = w0;
  *(uint4*)(Wbf + idx + 8) = w1;
}

// Kernel 1: per-row sum of exp(30*dot - 30) over a column block.
__global__ __launch_bounds__(256, 1) void oim_gemm_expsum(
    const float* __restrict__ X, const __bf16* __restrict__ Wbf,
    float* __restrict__ Spart) {
  __shared__ __align__(16) __bf16 sB[2][16 * CSTR];

  const int tid  = threadIdx.x;
  const int wave = tid >> 5;
  const int lane = tid & 31;
  const int m    = lane & 15;     // row within tile / col within tile
  const int hi   = lane >> 4;     // half-wave select
  const int rowBase = blockIdx.x * ROWBLK + wave * 32;
  const int cb   = blockIdx.y;

  const int t0 = cb * TILES_PER_CB;
  const int t1 = min(t0 + TILES_PER_CB, NTILES);

  // Kick the first TDM before the (long-latency) A staging so they overlap.
  if (wave == 0)
    tdm_load_tile(Wbf + (size_t)t0 * 16 * NF,
                  (unsigned)(uintptr_t)&sB[0][0], NC - t0 * 16);

  // Two 16-row A tiles per wave, held in registers (128 VGPRs).
  v16bf A0[8], A1[8];
  load_A(A0, X + (size_t)(rowBase + m) * NF, hi);
  load_A(A1, X + (size_t)(rowBase + 16 + m) * NF, hi);

  if (wave == 0) __builtin_amdgcn_s_wait_tensorcnt(0);
  __syncthreads();

  v8f sum0 = {}, sum1 = {};

  for (int t = t0; t < t1; ++t) {
    const int buf = (t - t0) & 1;

    // DMA tile t+1 into the other buffer while we compute tile t.
    if (wave == 0 && t + 1 < t1)
      tdm_load_tile(Wbf + (size_t)(t + 1) * 16 * NF,
                    (unsigned)(uintptr_t)&sB[buf ^ 1][0], NC - (t + 1) * 16);

    // Compute tile t: each B chunk feeds two WMMAs (two accumulation chains).
    const __bf16* bp = &sB[buf][m * CSTR + hi * 8];
    v8f c0 = {}, c1 = {};
    #pragma unroll
    for (int kc = 0; kc < 8; ++kc) {
      const __bf16* q = bp + kc * 32;
      B32x8 raw;
      raw.a = *(const uint4*)(q);
      raw.b = *(const uint4*)(q + 16);
      v16bf b = __builtin_bit_cast(v16bf, raw);
      c0 = __builtin_amdgcn_wmma_f32_16x16x32_bf16(
               false, A0[kc], false, b, (short)0, c0, false, false);
      c1 = __builtin_amdgcn_wmma_f32_16x16x32_bf16(
               false, A1[kc], false, b, (short)0, c1, false, false);
    }

    // Streaming epilogue with fixed -30 shift; mask padded tail columns
    // (exp(-30) is NOT negligible vs typical exp(30d-30) terms).
    const int coln = t * 16 + m;
    if (coln < NC) {
      #pragma unroll
      for (int r = 0; r < 8; ++r) {
        sum0[r] += __builtin_amdgcn_exp2f(fmaf(c0[r], LOG2E30, -LOG2E30));
        sum1[r] += __builtin_amdgcn_exp2f(fmaf(c1[r], LOG2E30, -LOG2E30));
      }
    }

    // Drain the in-flight TDM (issuing wave only), then publish the buffer.
    if (wave == 0 && t + 1 < t1) __builtin_amdgcn_s_wait_tensorcnt(0);
    __syncthreads();
  }

  // Row sums: reduce across the 16 lanes of each half-wave.
  #pragma unroll
  for (int r = 0; r < 8; ++r) {
    float v0 = sum0[r], v1 = sum1[r];
    v0 += __shfl_xor(v0, 8, 16);  v1 += __shfl_xor(v1, 8, 16);
    v0 += __shfl_xor(v0, 4, 16);  v1 += __shfl_xor(v1, 4, 16);
    v0 += __shfl_xor(v0, 2, 16);  v1 += __shfl_xor(v1, 2, 16);
    v0 += __shfl_xor(v0, 1, 16);  v1 += __shfl_xor(v1, 1, 16);
    if (m == 0) {   // lanes 0 (rows r) and 16 (rows r+8)
      float* dst = Spart + (size_t)cb * NROWS + rowBase + hi * 8 + r;
      dst[0]  = v0;
      dst[16] = v1;
    }
  }
}

// Kernel 2: exact-fp32 label logit per row (one wave per row).
__global__ __launch_bounds__(256) void oim_label_logit(
    const float* __restrict__ X, const int* __restrict__ roi,
    const float* __restrict__ lut, const float* __restrict__ cq,
    float* __restrict__ L) {
  const int wave = threadIdx.x >> 5;
  const int lane = threadIdx.x & 31;
  const int row  = blockIdx.x * 8 + wave;
  const int lab  = roi[row] - 1;
  float res = 0.f;
  if (lab != (IGNORE_ROI - 1) && lab >= 0 && lab < NC) {
    const float* wr = (lab < NP) ? (lut + (size_t)lab * NF)
                                 : (cq + (size_t)(lab - NP) * NF);
    const float* xr = X + (size_t)row * NF;
    float acc = 0.f;
    #pragma unroll
    for (int i = 0; i < 8; ++i) {
      const int k = lane + i * 32;
      acc = fmaf(xr[k], wr[k], acc);
    }
    acc += __shfl_xor(acc, 16, 32);
    acc += __shfl_xor(acc, 8, 32);
    acc += __shfl_xor(acc, 4, 32);
    acc += __shfl_xor(acc, 2, 32);
    acc += __shfl_xor(acc, 1, 32);
    res = 30.0f * acc;
  }
  if (lane == 0) L[row] = res;
}

// Kernel 3: loss = mean over valid rows of (30 + ln(S_i) - L_i).
__global__ __launch_bounds__(256) void oim_finalize(
    const float* __restrict__ Spart, const float* __restrict__ L,
    const int* __restrict__ roi, float* __restrict__ out) {
  __shared__ double ssum[256];
  __shared__ int    scnt[256];
  double acc = 0.0;
  int cnt = 0;
  for (int i = threadIdx.x; i < NROWS; i += 256) {
    if (roi[i] != IGNORE_ROI) {
      float s = 0.f;
      #pragma unroll
      for (int cbi = 0; cbi < NCB; ++cbi) s += Spart[(size_t)cbi * NROWS + i];
      const float lse = 30.0f + __builtin_amdgcn_logf(s) * LN2;
      acc += (double)(lse - L[i]);
      ++cnt;
    }
  }
  ssum[threadIdx.x] = acc;
  scnt[threadIdx.x] = cnt;
  __syncthreads();
  for (int s = 128; s > 0; s >>= 1) {
    if (threadIdx.x < s) {
      ssum[threadIdx.x] += ssum[threadIdx.x + s];
      scnt[threadIdx.x] += scnt[threadIdx.x + s];
    }
    __syncthreads();
  }
  if (threadIdx.x == 0) {
    const int n = scnt[0] > 0 ? scnt[0] : 1;
    out[0] = (float)(ssum[0] / (double)n);
  }
}

extern "C" void kernel_launch(void* const* d_in, const int* in_sizes, int n_in,
                              void* d_out, int out_size, void* d_ws, size_t ws_size,
                              hipStream_t stream) {
  (void)in_sizes; (void)n_in; (void)out_size; (void)ws_size;
  const float* X   = (const float*)d_in[0];   // inputs   [8192 x 256]
  const int*   roi = (const int*)  d_in[1];   // roi_label[8192]
  const float* lut = (const float*)d_in[2];   // lut      [5532 x 256]
  const float* cq  = (const float*)d_in[3];   // cq       [5000 x 256]

  // Workspace: Wbf[NC*NF] bf16, then Spart[NCB][NROWS], then L[NROWS] (~5.7 MB).
  __bf16* Wbf  = (__bf16*)d_ws;
  float* Spart = (float*)((char*)d_ws + WBF_BYTES);
  float* Llog  = Spart + (size_t)NCB * NROWS;

  const int cvt_blocks = (int)(((size_t)NC * NF / 16 + 255) / 256);   // 659
  oim_cvt_w<<<cvt_blocks, 256, 0, stream>>>(lut, cq, Wbf);

  dim3 g1(NROWS / ROWBLK, NCB);                       // 32 x 8 workgroups
  oim_gemm_expsum<<<g1, 256, 0, stream>>>(X, Wbf, Spart);
  oim_label_logit<<<NROWS / 8, 256, 0, stream>>>(X, roi, lut, cq, Llog);
  oim_finalize<<<1, 256, 0, stream>>>(Spart, Llog, roi, (float*)d_out);
}
